// net_2310692405841
// MI455X (gfx1250) — compile-verified, matrix operand order checked by reading, not run on hardware
//
#include <hip/hip_runtime.h>
#include <hip/hip_bf16.h>
#include <math.h>

// ------------------------- model constants -------------------------
constexpr int BATCH  = 16;
constexpr int NTOK   = 577;         // 24*24 patches + cls
constexpr int NPAD   = 608;         // padded: 19*32 = 38*16
constexpr int DIMM   = 512;
constexpr int NH     = 8;
constexpr int HD     = 64;
constexpr int LLAY   = 10;
constexpr int MROWS  = BATCH * NPAD;   // 9728 = 76*128
#define SCALE_F 0.04419417382415922f   // 512^-0.5

typedef __bf16 bf16;
typedef bf16  v16bf __attribute__((ext_vector_type(16)));
typedef float v8f   __attribute__((ext_vector_type(8)));
typedef unsigned short u16;
typedef unsigned int   u32;

__device__ __forceinline__ u16 f2bf(float f) {
    u32 u = __float_as_uint(f);
    u32 r = (u + 0x7FFFu + ((u >> 16) & 1u)) >> 16;
    return (u16)r;
}

union Frag {
    v16bf v;
    u16   u[16];
    uint4 q[2];
};

__device__ __forceinline__ v8f wmma_bf16(const Frag& a, const Frag& b, v8f c) {
    return __builtin_amdgcn_wmma_f32_16x16x32_bf16(
        /*neg_a=*/false, a.v, /*neg_b=*/false, b.v,
        /*c_mod=*/(short)0, c, /*reuse_a=*/false, /*reuse_b=*/false);
}

__device__ __forceinline__ float gelu_exact(float x) {
    return 0.5f * x * (1.0f + erff(x * 0.70710678118654752f));
}

// ------------------------- f32 -> bf16 convert -------------------------
__global__ void f32_to_bf16(const float* __restrict__ src, u16* __restrict__ dst, int n) {
    int i = blockIdx.x * 256 + threadIdx.x;
    if (i < n) dst[i] = f2bf(src[i]);
}

// ------------------------- patch embedding + pos emb -------------------------
__global__ __launch_bounds__(128)
void patch_embed(const float* __restrict__ x, const float* __restrict__ pos,
                 const float* __restrict__ cls, const float* __restrict__ peW,
                 const float* __restrict__ peB, float* __restrict__ h) {
    int row = blockIdx.x;
    int b = row / NPAD;
    int n = row % NPAD;
    int tid = threadIdx.x;
    float* out = h + (size_t)row * DIMM;
    if (n >= NTOK) {
        #pragma unroll
        for (int i = 0; i < 4; ++i) out[tid + i * 128] = 0.f;
        return;
    }
    if (n == 0) {
        #pragma unroll
        for (int i = 0; i < 4; ++i) { int d = tid + i * 128; out[d] = cls[d] + pos[d]; }
        return;
    }
    __shared__ float pv[48];
    int p = n - 1, gy = p / 24, gx = p % 24;
    if (tid < 48) {
        int py = tid / 12, rem = tid % 12, px = rem / 3, ch = rem % 3;
        pv[tid] = x[(((size_t)b * 3 + ch) * 96 + (gy * 4 + py)) * 96 + gx * 4 + px];
    }
    __syncthreads();
    #pragma unroll
    for (int i = 0; i < 4; ++i) {
        int d = tid + i * 128;
        float acc = peB[d] + pos[(size_t)n * DIMM + d];
        #pragma unroll 8
        for (int k = 0; k < 48; ++k) acc += pv[k] * peW[(size_t)k * DIMM + d];
        out[d] = acc;
    }
}

// ------------------------- layernorm -> bf16 -------------------------
__global__ __launch_bounds__(256)
void ln_bf16(const float* __restrict__ h, const float* __restrict__ g,
             const float* __restrict__ b, u16* __restrict__ y) {
    int row = blockIdx.x;
    int n = row % NPAD;
    int tid = threadIdx.x;
    if (n >= NTOK) {   // uniform per block
        y[(size_t)row * DIMM + tid] = 0;
        y[(size_t)row * DIMM + tid + 256] = 0;
        return;
    }
    __shared__ float red[256], red2[256];
    const float* xr = h + (size_t)row * DIMM;
    float a0 = xr[tid], a1 = xr[tid + 256];
    red[tid] = a0 + a1; red2[tid] = a0 * a0 + a1 * a1;
    __syncthreads();
    for (int s = 128; s > 0; s >>= 1) {
        if (tid < s) { red[tid] += red[tid + s]; red2[tid] += red2[tid + s]; }
        __syncthreads();
    }
    float mean = red[0] * (1.0f / 512.0f);
    float var  = red2[0] * (1.0f / 512.0f) - mean * mean;
    float rstd = rsqrtf(var + 1e-5f);
    y[(size_t)row * DIMM + tid]       = f2bf((a0 - mean) * rstd * g[tid] + b[tid]);
    y[(size_t)row * DIMM + tid + 256] = f2bf((a1 - mean) * rstd * g[tid + 256] + b[tid + 256]);
}

// ------------------------- tiled WMMA GEMM -------------------------
// C(MxN) = A(MxK,bf16) * Bw(KxN,bf16); 128x128 tile per 256-thread block (8 waves).
// Wave (wr,wc) owns a 32x64 sub-tile: 2 A-frags x 4 B-frags = 8 WMMA per K-chunk,
// each B-fragment reused across both A-fragments.
// mode 0: store bf16 (no bias)        -> Cb
// mode 1: + bias + resid, store f32   -> Cf
// mode 2: + bias, GELU, store bf16    -> Cb
__global__ __launch_bounds__(256)
void gemm_wmma(const u16* __restrict__ A, const u16* __restrict__ Bw,
               int M, int N, int K,
               float* __restrict__ Cf, u16* __restrict__ Cb,
               const float* __restrict__ bias, const float* __restrict__ resid,
               int mode) {
    __shared__ u16 As[128 * 32];   // [m][k]
    __shared__ u16 Bs[128 * 32];   // [n][k]  (transposed stage)
    const int tid  = threadIdx.x;
    const int lane = tid & 31;
    const int wave = tid >> 5;
    const int wr   = wave & 3;     // M 32-row block within tile
    const int wc   = wave >> 2;    // N 64-col block within tile
    const int m0 = blockIdx.y * 128;
    const int n0 = blockIdx.x * 128;

    v8f acc[2][4];
    #pragma unroll
    for (int i = 0; i < 2; ++i)
        #pragma unroll
        for (int t = 0; t < 4; ++t)
            #pragma unroll
            for (int r = 0; r < 8; ++r) acc[i][t][r] = 0.f;

    const int colb = lane & 15;
    const int koff = (lane < 16) ? 0 : 8;

    for (int kc = 0; kc < K; kc += 32) {
        // prefetch next K-chunk (gfx1250 global_prefetch_b8)
        if (kc + 32 < K) {
            int pr = tid >> 1;  // 0..127
            __builtin_prefetch(A + (size_t)(m0 + pr) * K + kc + 32 + (tid & 1) * 16, 0, 1);
            if (tid < 64)
                __builtin_prefetch(Bw + (size_t)(kc + 32 + (tid & 31)) * N + n0 + (tid >> 5) * 64, 0, 1);
        }
        // stage A tile: 128x32, 512 chunks of 8 halves, 2 per thread
        #pragma unroll
        for (int i = 0; i < 2; ++i) {
            int ch = tid * 2 + i;
            int r  = ch >> 2;
            int co = (ch & 3) * 8;
            *(uint4*)(&As[r * 32 + co]) =
                *(const uint4*)(A + (size_t)(m0 + r) * K + kc + co);
        }
        // stage B tile transposed into Bs[n][k]: 32k x 128n
        #pragma unroll
        for (int i = 0; i < 2; ++i) {
            int ch = tid * 2 + i;
            int kk = ch >> 4;            // 0..31
            int nb = (ch & 15) * 8;      // 0..120
            u16 tmp[8];
            *(uint4*)tmp = *(const uint4*)(Bw + (size_t)(kc + kk) * N + n0 + nb);
            #pragma unroll
            for (int t = 0; t < 8; ++t) Bs[(nb + t) * 32 + kk] = tmp[t];
        }
        __syncthreads();

        Frag af[2];
        #pragma unroll
        for (int i = 0; i < 2; ++i) {
            int ar = wr * 32 + i * 16 + colb;
            af[i].q[0] = *(const uint4*)(&As[ar * 32 + koff]);
            af[i].q[1] = *(const uint4*)(&As[ar * 32 + 16 + koff]);
        }
        #pragma unroll
        for (int t = 0; t < 4; ++t) {
            Frag bfr;
            int br = wc * 64 + t * 16 + colb;
            bfr.q[0] = *(const uint4*)(&Bs[br * 32 + koff]);
            bfr.q[1] = *(const uint4*)(&Bs[br * 32 + 16 + koff]);
            acc[0][t] = wmma_bf16(af[0], bfr, acc[0][t]);
            acc[1][t] = wmma_bf16(af[1], bfr, acc[1][t]);
        }
        __syncthreads();
    }

    const int half = lane >> 4;
    #pragma unroll
    for (int i = 0; i < 2; ++i) {
        #pragma unroll
        for (int t = 0; t < 4; ++t) {
            int n = n0 + wc * 64 + t * 16 + colb;
            #pragma unroll
            for (int r = 0; r < 8; ++r) {
                size_t m = (size_t)(m0 + wr * 32 + i * 16 + half * 8 + r);
                float v = acc[i][t][r];
                if (mode == 0) {
                    Cb[m * N + n] = f2bf(v);
                } else if (mode == 1) {
                    Cf[m * N + n] = v + bias[n] + resid[m * N + n];
                } else {
                    Cb[m * N + n] = f2bf(gelu_exact(v + bias[n]));
                }
            }
        }
    }
}

// ------------------------- flash attention -------------------------
// 64 threads = 2 waves per block; each wave owns one 16-row q-tile of the same
// (b, head); the V-transpose stage in LDS is shared between the two waves.
__global__ __launch_bounds__(64)
void attention(const u16* __restrict__ qkv, u16* __restrict__ O) {
    __shared__ u16 Ps[2][16 * 32];   // per-wave P (bf16, row-major)
    __shared__ u16 Vt[64 * 32];      // shared [d][j] V transpose
    const int QT2 = NPAD / 32;       // 19 q-tile pairs
    int wid = blockIdx.x;
    int qp = wid % QT2;
    int hh = (wid / QT2) % NH;
    int b  = wid / (QT2 * NH);
    int tid  = threadIdx.x;
    int lane = tid & 31;
    int wave = tid >> 5;
    int qt   = qp * 2 + wave;
    int col  = lane & 15;
    int half = lane >> 4;
    int koff = half ? 8 : 0;

    const size_t rstride = 3 * DIMM;  // 1536

    // Q fragments (A-operand) straight from global
    Frag qf[2];
    size_t baseQ = ((size_t)b * NPAD + qt * 16 + col) * rstride + hh * HD;
    #pragma unroll
    for (int c = 0; c < 2; ++c) {
        qf[c].q[0] = *(const uint4*)(qkv + baseQ + c * 32 + koff);
        qf[c].q[1] = *(const uint4*)(qkv + baseQ + c * 32 + 16 + koff);
    }

    v8f acc[4];
    float mrow[8], lrow[8];
    #pragma unroll
    for (int t = 0; t < 4; ++t)
        #pragma unroll
        for (int r = 0; r < 8; ++r) acc[t][r] = 0.f;
    #pragma unroll
    for (int r = 0; r < 8; ++r) { mrow[r] = -1e30f; lrow[r] = 0.f; }

    for (int j0 = 0; j0 < NPAD; j0 += 32) {
        // ---- S = Q K^T for 2 j-tiles (K rows are exactly the B-operand layout)
        v8f s[2];
        #pragma unroll
        for (int jt = 0; jt < 2; ++jt) {
            size_t baseK = ((size_t)b * NPAD + j0 + jt * 16 + col) * rstride + DIMM + hh * HD;
            v8f sv;
            #pragma unroll
            for (int r = 0; r < 8; ++r) sv[r] = 0.f;
            #pragma unroll
            for (int c = 0; c < 2; ++c) {
                Frag kf;
                kf.q[0] = *(const uint4*)(qkv + baseK + c * 32 + koff);
                kf.q[1] = *(const uint4*)(qkv + baseK + c * 32 + 16 + koff);
                sv = wmma_bf16(qf[c], kf, sv);
            }
            int j = j0 + jt * 16 + col;
            float msk = (j < NTOK) ? 0.f : -1e30f;
            #pragma unroll
            for (int r = 0; r < 8; ++r) s[jt][r] = sv[r] * SCALE_F + msk;
        }
        // ---- cooperative V-transpose stage: 64 threads cover 32 j-rows x 64 d
        {
            int jl = tid & 31;        // j row
            int dh = tid >> 5;        // d half (0: d 0..31, 1: d 32..63)
            size_t baseV = ((size_t)b * NPAD + j0 + jl) * rstride + 2 * DIMM + hh * HD + dh * 32;
            #pragma unroll
            for (int dq = 0; dq < 4; ++dq) {
                uint4 v4 = *(const uint4*)(qkv + baseV + dq * 8);
                const u16* vp = (const u16*)&v4;
                #pragma unroll
                for (int t = 0; t < 8; ++t) Vt[(dh * 32 + dq * 8 + t) * 32 + jl] = vp[t];
            }
        }
        // ---- online softmax (row-wise, 16-lane reductions)
        float pnew[2][8];
        #pragma unroll
        for (int r = 0; r < 8; ++r) {
            float mx = fmaxf(s[0][r], s[1][r]);
            #pragma unroll
            for (int off = 8; off > 0; off >>= 1) mx = fmaxf(mx, __shfl_xor(mx, off, 32));
            float mnew = fmaxf(mrow[r], mx);
            float alpha = __expf(mrow[r] - mnew);
            float p0 = __expf(s[0][r] - mnew);
            float p1 = __expf(s[1][r] - mnew);
            float rs = p0 + p1;
            #pragma unroll
            for (int off = 8; off > 0; off >>= 1) rs += __shfl_xor(rs, off, 32);
            lrow[r] = lrow[r] * alpha + rs;
            mrow[r] = mnew;
            #pragma unroll
            for (int t = 0; t < 4; ++t) acc[t][r] *= alpha;
            pnew[0][r] = p0; pnew[1][r] = p1;
        }
        // ---- P: C-layout -> LDS row-major
        #pragma unroll
        for (int jt = 0; jt < 2; ++jt)
            #pragma unroll
            for (int r = 0; r < 8; ++r)
                Ps[wave][(half * 8 + r) * 32 + jt * 16 + col] = f2bf(pnew[jt][r]);
        __syncthreads();

        // ---- O += P V
        Frag pf;
        pf.q[0] = *(const uint4*)(&Ps[wave][col * 32 + koff]);
        pf.q[1] = *(const uint4*)(&Ps[wave][col * 32 + 16 + koff]);
        #pragma unroll
        for (int t = 0; t < 4; ++t) {
            Frag vf;
            int vr = t * 16 + col;
            vf.q[0] = *(const uint4*)(&Vt[vr * 32 + koff]);
            vf.q[1] = *(const uint4*)(&Vt[vr * 32 + 16 + koff]);
            acc[t] = wmma_bf16(pf, vf, acc[t]);
        }
        __syncthreads();
    }

    // ---- O = acc / l, store bf16
    #pragma unroll
    for (int t = 0; t < 4; ++t)
        #pragma unroll
        for (int r = 0; r < 8; ++r) {
            float l = lrow[r];
            float o = (l > 0.f) ? acc[t][r] / l : 0.f;
            size_t m = (size_t)b * NPAD + qt * 16 + half * 8 + r;
            O[m * DIMM + hh * HD + t * 16 + col] = f2bf(o);
        }
}

// ------------------------- classifier head -------------------------
__global__ __launch_bounds__(512)
void head_kernel(const float* __restrict__ h, const float* __restrict__ hg,
                 const float* __restrict__ hb, const float* __restrict__ w1,
                 const float* __restrict__ b1, const float* __restrict__ w2,
                 const float* __restrict__ b2, float* __restrict__ out) {
    int b = blockIdx.x;
    int tid = threadIdx.x;
    __shared__ float zn[512], z1[512], red[512], red2[512];
    const float* xr = h + (size_t)b * NPAD * DIMM;   // cls token row
    float v = xr[tid];
    red[tid] = v; red2[tid] = v * v;
    __syncthreads();
    for (int s = 256; s > 0; s >>= 1) {
        if (tid < s) { red[tid] += red[tid + s]; red2[tid] += red2[tid + s]; }
        __syncthreads();
    }
    float mean = red[0] * (1.0f / 512.0f);
    float var  = red2[0] * (1.0f / 512.0f) - mean * mean;
    float rstd = rsqrtf(var + 1e-5f);
    zn[tid] = (v - mean) * rstd * hg[tid] + hb[tid];
    __syncthreads();
    float a = b1[tid];
    for (int k = 0; k < 512; ++k) a += zn[k] * w1[(size_t)k * 512 + tid];
    z1[tid] = gelu_exact(a);
    __syncthreads();
    if (tid < 10) {
        float a2 = b2[tid];
        for (int k = 0; k < 512; ++k) a2 += z1[k] * w2[(size_t)k * 10 + tid];
        out[b * 10 + tid] = a2;
    }
}

// ------------------------- host orchestration -------------------------
extern "C" void kernel_launch(void* const* d_in, const int* in_sizes, int n_in,
                              void* d_out, int out_size, void* d_ws, size_t ws_size,
                              hipStream_t stream) {
    (void)in_sizes; (void)n_in; (void)out_size; (void)ws_size;
    const float* x      = (const float*)d_in[0];
    const float* pos    = (const float*)d_in[1];
    const float* cls    = (const float*)d_in[2];
    const float* peW    = (const float*)d_in[3];
    const float* peB    = (const float*)d_in[4];
    const float* ln1_g  = (const float*)d_in[5];
    const float* ln1_b  = (const float*)d_in[6];
    const float* qkv_W  = (const float*)d_in[7];
    const float* out_W  = (const float*)d_in[8];
    const float* out_b  = (const float*)d_in[9];
    const float* ln2_g  = (const float*)d_in[10];
    const float* ln2_b  = (const float*)d_in[11];
    const float* ff1_W  = (const float*)d_in[12];
    const float* ff1_b  = (const float*)d_in[13];
    const float* ff2_W  = (const float*)d_in[14];
    const float* ff2_b  = (const float*)d_in[15];
    const float* hln_g  = (const float*)d_in[16];
    const float* hln_b  = (const float*)d_in[17];
    const float* h1_W   = (const float*)d_in[18];
    const float* h1_b   = (const float*)d_in[19];
    const float* h2_W   = (const float*)d_in[20];
    const float* h2_b   = (const float*)d_in[21];

    char* ws = (char*)d_ws;
    size_t off = 0;
    auto alloc = [&](size_t bytes) -> void* {
        void* p = ws + off;
        off = (off + bytes + 255) & ~(size_t)255;
        return p;
    };
    u16*   wqkv = (u16*)alloc((size_t)LLAY * 512 * 1536 * 2);
    u16*   wout = (u16*)alloc((size_t)LLAY * 512 * 512 * 2);
    u16*   wff1 = (u16*)alloc((size_t)LLAY * 512 * 512 * 2);
    u16*   wff2 = (u16*)alloc((size_t)LLAY * 512 * 512 * 2);
    float* h    = (float*)alloc((size_t)MROWS * DIMM * 4);
    u16*   y    = (u16*)alloc((size_t)MROWS * DIMM * 2);
    u16*   qkvb = (u16*)alloc((size_t)MROWS * 3 * DIMM * 2);
    u16*   ob   = (u16*)alloc((size_t)MROWS * DIMM * 2);
    u16*   act  = (u16*)alloc((size_t)MROWS * DIMM * 2);

    // weight conversion (once per launch; deterministic)
    int n;
    n = LLAY * 512 * 1536;
    f32_to_bf16<<<(n + 255) / 256, 256, 0, stream>>>(qkv_W, wqkv, n);
    n = LLAY * 512 * 512;
    f32_to_bf16<<<(n + 255) / 256, 256, 0, stream>>>(out_W, wout, n);
    f32_to_bf16<<<(n + 255) / 256, 256, 0, stream>>>(ff1_W, wff1, n);
    f32_to_bf16<<<(n + 255) / 256, 256, 0, stream>>>(ff2_W, wff2, n);

    patch_embed<<<MROWS, 128, 0, stream>>>(x, pos, cls, peW, peB, h);

    dim3 gQKV(3 * DIMM / 128, MROWS / 128);
    dim3 gSQ(DIMM / 128, MROWS / 128);
    for (int l = 0; l < LLAY; ++l) {
        ln_bf16<<<MROWS, 256, 0, stream>>>(h, ln1_g + l * 512, ln1_b + l * 512, y);
        gemm_wmma<<<gQKV, 256, 0, stream>>>(y, wqkv + (size_t)l * 512 * 1536,
                                            MROWS, 3 * DIMM, DIMM,
                                            nullptr, qkvb, nullptr, nullptr, 0);
        attention<<<BATCH * NH * (NPAD / 32), 64, 0, stream>>>(qkvb, ob);
        gemm_wmma<<<gSQ, 256, 0, stream>>>(ob, wout + (size_t)l * 512 * 512,
                                           MROWS, DIMM, DIMM,
                                           h, nullptr, out_b + l * 512, h, 1);
        ln_bf16<<<MROWS, 256, 0, stream>>>(h, ln2_g + l * 512, ln2_b + l * 512, y);
        gemm_wmma<<<gSQ, 256, 0, stream>>>(y, wff1 + (size_t)l * 512 * 512,
                                           MROWS, DIMM, DIMM,
                                           nullptr, act, ff1_b + l * 512, nullptr, 2);
        gemm_wmma<<<gSQ, 256, 0, stream>>>(act, wff2 + (size_t)l * 512 * 512,
                                           MROWS, DIMM, DIMM,
                                           h, nullptr, ff2_b + l * 512, h, 1);
    }

    head_kernel<<<BATCH, 512, 0, stream>>>(h, hln_g, hln_b, h1_W, h1_b, h2_W, h2_b,
                                           (float*)d_out);
}
